// ModuleCorrelation_44195213476622
// MI455X (gfx1250) — matrix operand order repaired and checked by reading, not compiled, not run on hardware
//
#include <hip/hip_runtime.h>

// Problem constants (fixed by the reference harness)
#define BB 8
#define CC 128
#define HH 192
#define WW 192
#define NDISP 9   // per axis; 81 output channels

typedef __attribute__((ext_vector_type(16))) __bf16 v16bf;
typedef __attribute__((ext_vector_type(8)))  float  v8f;

union Frag {
    uint4 q[2];
    v16bf v;
};

// round-to-nearest-even f32 -> bf16, packed pair (f1 in high half)
static __device__ inline unsigned int pack2bf(float f0, float f1) {
    unsigned int b0 = __float_as_uint(f0);
    unsigned int b1 = __float_as_uint(f1);
    unsigned int r0 = (b0 + 0x7FFFu + ((b0 >> 16) & 1u)) >> 16;
    unsigned int r1 = (b1 + 0x7FFFu + ((b1 >> 16) & 1u)) & 0xFFFF0000u;
    return r1 | r0;
}

__global__ __launch_bounds__(32)
void corr_wmma_kernel(const float* __restrict__ first,
                      const float* __restrict__ second,
                      float* __restrict__ out) {
    // LDS staging, padded rows: 136 ushorts = 272B (16B aligned, bank-spread)
    __shared__ unsigned short Af[16][136];  // first tile  [x][ch]  bf16
    __shared__ unsigned short Bs[32][136];  // second tile [x'][ch] bf16 (x' = x0-4 .. x0+27)
    __shared__ float          Gs[16][28];   // G band stage [m][col]

    const int lane = threadIdx.x;           // 0..31 (wave32)
    const int b    = blockIdx.z;
    const int y    = blockIdx.y;
    const int x0   = blockIdx.x << 4;

    const long chStride = (long)HH * WW;

    // ---------------- load & convert first tile: 16 x * 128 ch -----------------
    // lanes 0-15 handle ch c8+0..3, lanes 16-31 handle ch c8+4..7 (4 ch per lane)
    {
        const int colA  = lane & 15;
        const int chOff = (lane >> 4) << 2;   // 0 or 4
        const float* fb = first + (((long)b * CC) * HH + y) * WW + (x0 + colA);
        for (int c8 = 0; c8 < CC; c8 += 8) {
            int ch = c8 + chOff;
            const float* p = fb + (long)ch * chStride;
            float f0 = p[0];
            float f1 = p[chStride];
            float f2 = p[2 * chStride];
            float f3 = p[3 * chStride];
            uint2 w;
            w.x = pack2bf(f0, f1);
            w.y = pack2bf(f2, f3);
            *(uint2*)&Af[colA][ch] = w;
        }
    }
    __syncthreads();

    // ---------------- build A fragments (kept in VGPRs across all dy) ----------
    // 16-bit A 16x32 layout: lanes 0-15 hold M=lane, K = k0..k0+7 then k0+16..k0+23
    //                        lanes 16-31 hold M=lane-16, K = k0+8..k0+15 then k0+24..k0+31
    Frag afr[4];
    {
        const int m  = lane & 15;
        const int hi = (lane >> 4) << 3;  // +8 halves for hi lanes
        for (int kc = 0; kc < 4; ++kc) {
            int ko = kc * 32 + hi;
            afr[kc].q[0] = *(const uint4*)&Af[m][ko];
            afr[kc].q[1] = *(const uint4*)&Af[m][ko + 16];
        }
    }

    const int  n     = lane & 15;
    const int  koHi  = (lane >> 4) << 3;   // B: lanes 16-31 hold K = k0+8..k0+15 then k0+24..k0+31
    const int  xp    = x0 - 4 + lane;      // lane -> x' column of second tile
    const bool colOk = (unsigned)xp < (unsigned)WW;
    const int  xc    = colOk ? xp : 0;
    const float scale = 1.0f / (float)CC;

    // ---------------- loop over dy: 9 rows of second --------------------------
    for (int dy = 0; dy < NDISP; ++dy) {
        const int  yq    = y + dy - 4;
        const bool rowOk = (unsigned)yq < (unsigned)HH;
        const int  yc    = rowOk ? yq : 0;
        const bool ok    = rowOk && colOk;

        // Branch-free fill: addresses are clamped in-range, zero via select.
        const float* sb = second + (((long)b * CC) * HH + yc) * WW + xc;
        for (int c4 = 0; c4 < CC; c4 += 4) {
            const float* p = sb + (long)c4 * chStride;
            float f0 = p[0];
            float f1 = p[chStride];
            float f2 = p[2 * chStride];
            float f3 = p[3 * chStride];
            f0 = ok ? f0 : 0.0f;
            f1 = ok ? f1 : 0.0f;
            f2 = ok ? f2 : 0.0f;
            f3 = ok ? f3 : 0.0f;
            uint2 w;
            w.x = pack2bf(f0, f1);
            w.y = pack2bf(f2, f3);
            *(uint2*)&Bs[lane][c4] = w;
        }

        // Warm L2/WGP$ for the next dy row (speculative, clamped addresses).
        if (dy < NDISP - 1) {
            int yn = yq + 1;
            if (yn > HH - 1) yn = HH - 1;
            if (yn < 0) yn = 0;
            #pragma unroll
            for (int q = 0; q < 4; ++q) {
                int ch = q * 32 + lane;
                __builtin_prefetch(second + (((long)b * CC + ch) * HH + yn) * WW + xc, 0, 3);
            }
        }
        __syncthreads();

        // G = A(16x128) * B(128x32) via 4 K-steps x 2 N-tiles of bf16 WMMA
        v8f acc0 = {};
        v8f acc1 = {};
        for (int kc = 0; kc < 4; ++kc) {
            int k0 = kc * 32 + koHi;
            Frag bf0, bf1;
            bf0.q[0] = *(const uint4*)&Bs[n][k0];
            bf0.q[1] = *(const uint4*)&Bs[n][k0 + 16];
            bf1.q[0] = *(const uint4*)&Bs[n + 16][k0];
            bf1.q[1] = *(const uint4*)&Bs[n + 16][k0 + 16];
            acc0 = __builtin_amdgcn_wmma_f32_16x16x32_bf16(
                false, afr[kc].v, false, bf0.v, (short)0, acc0, false, false);
            acc1 = __builtin_amdgcn_wmma_f32_16x16x32_bf16(
                false, afr[kc].v, false, bf1.v, (short)0, acc1, false, false);
        }

        // stage G band to LDS: C/D layout -> VGPR v holds M = v (+8 for hi lanes), N = lane&15
        {
            const int rowBase = (lane >> 4) << 3;
            #pragma unroll
            for (int v = 0; v < 8; ++v) {
                Gs[rowBase + v][n]      = acc0[v];
                Gs[rowBase + v][16 + n] = acc1[v];
            }
        }
        __syncthreads();

        // extract out[dx, x] = G[m, m+dx'] ; 9*16 = 144 values, coalesced stores
        #pragma unroll
        for (int i = 0; i < 5; ++i) {
            int idx = i * 32 + lane;
            if (idx < 144) {
                int dxp = idx >> 4;
                int m   = idx & 15;
                float vv = Gs[m][m + dxp] * scale;
                out[(((long)b * 81 + dy * NDISP + dxp) * HH + y) * WW + (x0 + m)] = vv;
            }
        }
        __syncthreads();
    }
}

extern "C" void kernel_launch(void* const* d_in, const int* in_sizes, int n_in,
                              void* d_out, int out_size, void* d_ws, size_t ws_size,
                              hipStream_t stream) {
    (void)in_sizes; (void)n_in; (void)out_size; (void)d_ws; (void)ws_size;
    const float* first  = (const float*)d_in[0];
    const float* second = (const float*)d_in[1];
    float* out = (float*)d_out;

    dim3 grid(WW / 16, HH, BB);   // (12, 192, 8) workgroups, one wave each
    dim3 block(32, 1, 1);
    corr_wmma_kernel<<<grid, block, 0, stream>>>(first, second, out);
}
